// CategoricalDQNModel_28793460752482
// MI455X (gfx1250) — matrix operand order repaired and checked by reading, not compiled, not run on hardware
//
#include <hip/hip_runtime.h>
#include <hip/hip_bf16.h>

// ---------------------------------------------------------------------------
// C51 (categorical DQN) loss on MI455X / gfx1250.
// Two bf16 WMMA GEMMs ([B,512]x[512,918]) + fused softmax/argmax/Bellman
// projection epilogues. f32 accumulate via v_wmma_f32_16x16x32_bf16.
// ---------------------------------------------------------------------------

typedef __attribute__((ext_vector_type(16))) __bf16 v16bf;
typedef __attribute__((ext_vector_type(8)))  float  v8f;

#define NH     512          // hidden
#define NA     18           // actions
#define NAT    51           // atoms
#define NPAD   64           // atoms padded to multiple of 16
#define NCOLS  (NA * NPAD)  // 1152 padded output columns
#define LGROW  1156         // padded LDS row stride (floats) for logits
#define XROW   520          // padded LDS row stride (halfs) for x tile
#define WELEMS (NA * NPAD * NH) // 589824 bf16 elems per converted weight

__device__ __forceinline__ unsigned short f32_to_bf16(float f) {
    unsigned int u = __float_as_uint(f);
    unsigned int r = (u + 0x7FFFu + ((u >> 16) & 1u)) >> 16;   // RNE
    return (unsigned short)r;
}

union U16 {
    uint4 q[2];
    v16bf v;
};

// ---------------------------------------------------------------------------
// Kernel 0: convert W [H][A][51] f32 -> [A][64][H] bf16 (column-major per
// output column, zero-padded atoms 51..63) for both heads.
// ---------------------------------------------------------------------------
__global__ __launch_bounds__(256) void convert_weights(
    const float* __restrict__ W, const float* __restrict__ Wt,
    unsigned short* __restrict__ W16, unsigned short* __restrict__ Wt16) {
    int idx = blockIdx.x * 256 + threadIdx.x;
    if (idx >= 2 * WELEMS) return;
    int which = (idx >= WELEMS);
    int e = which ? idx - WELEMS : idx;
    int a = e >> 15;          // / (64*512)
    int n = (e >> 9) & 63;    // atom (padded)
    int k = e & 511;          // hidden index
    const float* src = which ? Wt : W;
    float v = (n < NAT) ? src[k * (NA * NAT) + a * NAT + n] : 0.0f;
    (which ? Wt16 : W16)[e] = f32_to_bf16(v);
}

// ---------------------------------------------------------------------------
// GEMM + epilogue. One workgroup = 16 rows of B, 8 waves x 9 column tiles
// = 72 tiles of 16 cols = 1152 padded cols. K = 512 in 16 steps of 32.
// TARGET=1: target head -> projected distribution m[B,51] into workspace.
// TARGET=0: train head  -> cross-entropy loss[B] into d_out.
// ---------------------------------------------------------------------------
template <int TARGET>
__global__ __launch_bounds__(256) void head_kernel(
    const float* __restrict__ xin,          // [B,512] activations
    const unsigned short* __restrict__ w16, // [A][64][512] bf16 weights
    const float* __restrict__ bias,         // [A][51]
    const float* __restrict__ reward,       // [B]      (target)
    const int*   __restrict__ action,       // [B]      (train)
    const int*   __restrict__ terminal,     // [B]      (target)
    const float* __restrict__ m_in,         // [B,51]   (train)
    float*       __restrict__ m_out,        // [B,51]   (target)
    float*       __restrict__ loss_out) {   // [B]      (train)

    __shared__ unsigned short xt[16 * XROW];   // bf16 activation tile
    __shared__ float lg[16 * LGROW];           // f32 logits tile
    __shared__ float qv[16 * NA];              // expected values
    __shared__ float mb[16 * 52];              // projection scratch

    const int tid   = threadIdx.x;
    const int bbase = blockIdx.x * 16;

    // ---- stage 16x512 f32 rows -> bf16 LDS tile (coalesced) ----
    for (int idx = tid; idx < 16 * NH; idx += 256) {
        int r = idx >> 9, c = idx & 511;
        xt[r * XROW + c] = f32_to_bf16(xin[(bbase + r) * NH + c]);
    }
    __syncthreads();

    // ---- WMMA main loop ----
    const int w    = tid >> 5;
    const int lane = tid & 31;
    const int mrow = lane & 15;
    const int sel8 = (lane < 16) ? 0 : 8;    // A-fragment K interleave
    const int selB = (lane < 16) ? 0 : 16;   // unused; B fragment also uses sel8

    v8f acc[9] = {};

    for (int ks = 0; ks < 16; ++ks) {
        const int Kb = ks * 32;
        U16 av;
        av.q[0] = *(const uint4*)&xt[mrow * XROW + Kb + sel8];
        av.q[1] = *(const uint4*)&xt[mrow * XROW + Kb + 16 + sel8];
#pragma unroll
        for (int t = 0; t < 9; ++t) {
            const int col = (w * 9 + t) * 16 + mrow;
            U16 bv;
            const uint4* pb = (const uint4*)&w16[col * NH + Kb + sel8];
            bv.q[0] = pb[0];
            bv.q[1] = pb[2];
            acc[t] = __builtin_amdgcn_wmma_f32_16x16x32_bf16(
                false, av.v, false, bv.v, (short)0, acc[t], false, false);
        }
    }

    // ---- accumulators (+bias) -> LDS logits tile ----
    const int moff = (lane < 16) ? 0 : 8;
#pragma unroll
    for (int t = 0; t < 9; ++t) {
        const int col = (w * 9 + t) * 16 + mrow;
        const int a = col >> 6, n = col & 63;
        const float bval = (n < NAT) ? bias[a * NAT + n] : 0.0f;
#pragma unroll
        for (int r = 0; r < 8; ++r) {
            lg[(r + moff) * LGROW + col] = acc[t][r] + bval;
        }
    }
    __syncthreads();

    if (TARGET) {
        // ---- expected value Q[r][a] = sum softmax(logits)*z ----
        for (int p = tid; p < 16 * NA; p += 256) {
            const int r = p / NA, a = p % NA;
            const float* row = &lg[r * LGROW + a * NPAD];
            float mx = row[0];
            for (int n = 1; n < NAT; ++n) mx = fmaxf(mx, row[n]);
            float s = 0.0f, qs = 0.0f;
            for (int n = 0; n < NAT; ++n) {
                float e = __expf(row[n] - mx);
                s += e;
                qs += e * (-10.0f + 0.4f * (float)n);
            }
            qv[p] = qs / s;
        }
        __syncthreads();
        // ---- greedy action, probs, Bellman projection ----
        if (tid < 16) {
            const int r = tid, b = bbase + r;
            int best = 0;
            float bq_ = qv[r * NA];
            for (int a = 1; a < NA; ++a) {
                float q = qv[r * NA + a];
                if (q > bq_) { bq_ = q; best = a; }
            }
            const float* row = &lg[r * LGROW + best * NPAD];
            float mx = row[0];
            for (int n = 1; n < NAT; ++n) mx = fmaxf(mx, row[n]);
            float s = 0.0f;
            for (int n = 0; n < NAT; ++n) s += __expf(row[n] - mx);
            const float inv_s = 1.0f / s;
            float* mrowp = &mb[r * 52];
            for (int n = 0; n < 52; ++n) mrowp[n] = 0.0f;
            const float rw = reward[b];
            const float notterm = 1.0f - (float)terminal[b];
            for (int n = 0; n < NAT; ++n) {
                const float p  = __expf(row[n] - mx) * inv_s;
                const float z  = -10.0f + 0.4f * (float)n;
                float Tz = rw + notterm * (z * 0.99f);
                Tz = fminf(10.0f, fmaxf(-10.0f, Tz));
                const float bq = (Tz + 10.0f) * 2.5f;   // / dz, dz = 0.4
                const float lo = floorf(bq), hi = ceilf(bq);
                mrowp[(int)lo] += p * (bq - lo);
                mrowp[(int)hi] += p * (hi - bq);
            }
            for (int n = 0; n < NAT; ++n) m_out[b * NAT + n] = mrowp[n];
        }
    } else {
        // ---- gather taken-action row, log-softmax, CE vs m ----
        if (tid < 16) {
            const int r = tid, b = bbase + r;
            const int a = action[b];
            const float* row = &lg[r * LGROW + a * NPAD];
            float mx = row[0];
            for (int n = 1; n < NAT; ++n) mx = fmaxf(mx, row[n]);
            float s = 0.0f;
            for (int n = 0; n < NAT; ++n) s += __expf(row[n] - mx);
            const float lse = __logf(s);
            float accum = 0.0f;
            for (int n = 0; n < NAT; ++n)
                accum += m_in[b * NAT + n] * (row[n] - mx - lse);
            loss_out[b] = -accum;
        }
    }
}

extern "C" void kernel_launch(void* const* d_in, const int* in_sizes, int n_in,
                              void* d_out, int out_size, void* d_ws, size_t ws_size,
                              hipStream_t stream) {
    const float* x        = (const float*)d_in[0];
    const float* next_x   = (const float*)d_in[1];
    const float* W        = (const float*)d_in[2];
    const float* b        = (const float*)d_in[3];
    const float* Wt       = (const float*)d_in[4];
    const float* bt       = (const float*)d_in[5];
    const float* reward   = (const float*)d_in[6];
    const int*   action   = (const int*)d_in[7];
    const int*   terminal = (const int*)d_in[8];
    float* out = (float*)d_out;

    const int B = in_sizes[6];          // batch from reward length

    // workspace layout
    char* ws = (char*)d_ws;
    unsigned short* W16  = (unsigned short*)ws;                       // 1,179,648 B
    unsigned short* Wt16 = (unsigned short*)(ws + (size_t)WELEMS * 2);
    float*          m    = (float*)(ws + (size_t)WELEMS * 4);         // B*51*4 B

    convert_weights<<<(2 * WELEMS + 255) / 256, 256, 0, stream>>>(W, Wt, W16, Wt16);

    const int grid = B / 16;
    head_kernel<1><<<grid, 256, 0, stream>>>(next_x, Wt16, bt, reward, nullptr,
                                             terminal, nullptr, m, nullptr);
    head_kernel<0><<<grid, 256, 0, stream>>>(x, W16, b, nullptr, action,
                                             nullptr, m, nullptr, out);
}
